// TopKAuto_12876311954115
// MI455X (gfx1250) — compile-verified, matrix operand order checked by reading, not run on hardware
//
#include <hip/hip_runtime.h>
#include <hip/hip_bf16.h>
#include <stdint.h>

#define B_ROWS 8192
#define D_DIM  1024
#define H_DIM  32768
#define TOPK   64
#define EPS_LN 1e-5f

typedef __attribute__((ext_vector_type(16))) _Float16 v16h;
typedef __attribute__((ext_vector_type(8)))  _Float16 v8h;
typedef __attribute__((ext_vector_type(4)))  _Float16 v4h;
typedef __attribute__((ext_vector_type(8)))  float    v8f;
typedef __attribute__((ext_vector_type(4)))  unsigned int u32x4;
typedef __attribute__((ext_vector_type(8)))  int      i32x8;
typedef __attribute__((ext_vector_type(4)))  int      i32x4;

// ---------------------------------------------------------------------------
// TDM: issue a 2D tensor_load_to_lds. D# packed per CDNA5 ISA 08_async_tensor:
//   group0: count=1, lds_addr, global_addr(57b), type=2
//   group1: data_size=1 (2B elems), tensor dims (large, no clipping),
//           tile_dim0=tile_w, tile_dim1=tile_h, tensor_dim0_stride=row stride
// Tile is written linearly to LDS: tile_h rows of tile_w contiguous elements.
// This toolchain exposes the 6-arg builtin:
//   (uint32x4 g0, int32x8 g1, int32x4 g2, int32x4 g3, int32x8 extra, i32 cpol)
// ---------------------------------------------------------------------------
__device__ __forceinline__ void tdm_load_tile_2d(uint32_t lds_byte_addr,
                                                 const void* gptr,
                                                 uint32_t tile_w, uint32_t tile_h,
                                                 uint64_t row_stride_elems)
{
    const uint64_t ga = (uint64_t)(uintptr_t)gptr;
    u32x4 g0;
    g0.x = 1u;                                              // count=1, user mode
    g0.y = lds_byte_addr;                                   // bits 63:32
    g0.z = (uint32_t)(ga & 0xFFFFFFFFull);                  // global_addr lo
    g0.w = (uint32_t)((ga >> 32) & 0x01FFFFFFull) | (2u << 30); // addr hi | type=2

    const uint64_t td0 = 1ull << 20;                        // tensor_dim0 (no clip)
    const uint64_t td1 = 1ull << 20;                        // tensor_dim1 (no clip)
    const uint64_t q0 = (1ull << 16)                        // data_size=1 (2 bytes)
                      | ((td0 & 0xFFFFull) << 48);          // tensor_dim0[15:0] @48
    const uint64_t q1 = ((td0 >> 16) & 0xFFFFull)           // tensor_dim0[31:16] @64
                      | ((td1 & 0xFFFFFFFFull) << 16)       // tensor_dim1 @80
                      | ((uint64_t)(tile_w & 0xFFFFu) << 48); // tile_dim0 @112
    const uint64_t q2 = (uint64_t)(tile_h & 0xFFFFu)        // tile_dim1 @128
                      | ((row_stride_elems & 0xFFFFFFFFull) << 32); // dim0_stride @160
    const uint64_t q3 = (row_stride_elems >> 32) & 0xFFFFull;       // stride hi @192

    i32x8 g1;
    g1[0] = (int)(uint32_t)q0;        g1[1] = (int)(uint32_t)(q0 >> 32);
    g1[2] = (int)(uint32_t)q1;        g1[3] = (int)(uint32_t)(q1 >> 32);
    g1[4] = (int)(uint32_t)q2;        g1[5] = (int)(uint32_t)(q2 >> 32);
    g1[6] = (int)(uint32_t)q3;        g1[7] = (int)(uint32_t)(q3 >> 32);
    i32x4 z4 = {0, 0, 0, 0};
    i32x8 z8 = {0, 0, 0, 0, 0, 0, 0, 0};
    __builtin_amdgcn_tensor_load_to_lds(g0, g1, z4, z4, z8, 0);
}

// ---------------------------------------------------------------------------
// Kernel 1: per-row LayerNorm, subtract b_dec, emit f16 A matrix + mean/std
// ---------------------------------------------------------------------------
__global__ __launch_bounds__(256)
void layernorm_kernel(const float* __restrict__ x, const float* __restrict__ b_dec,
                      _Float16* __restrict__ A_h, float* __restrict__ meanArr,
                      float* __restrict__ stdArr)
{
    const uint32_t row = blockIdx.x;
    const uint32_t tid = threadIdx.x;
    const float4 v = ((const float4*)(x + (size_t)row * D_DIM))[tid];

    float s  = v.x + v.y + v.z + v.w;
    float sq = v.x * v.x + v.y * v.y + v.z * v.z + v.w * v.w;

    __shared__ float rs[256], rq[256];
    rs[tid] = s; rq[tid] = sq;
    __syncthreads();
    #pragma unroll
    for (int off = 128; off > 0; off >>= 1) {
        if (tid < (uint32_t)off) { rs[tid] += rs[tid + off]; rq[tid] += rq[tid + off]; }
        __syncthreads();
    }
    __shared__ float sh_mean, sh_rinv;
    if (tid == 0) {
        const float mean = rs[0] * (1.0f / D_DIM);
        const float var  = fmaxf(rq[0] * (1.0f / D_DIM) - mean * mean, 0.0f);
        sh_mean = mean;
        sh_rinv = rsqrtf(var + EPS_LN);
        meanArr[row] = mean;
        stdArr[row]  = sqrtf(var);          // reference: std WITHOUT eps
    }
    __syncthreads();
    const float mean = sh_mean, rinv = sh_rinv;
    const float4 bd = ((const float4*)b_dec)[tid];

    v4h h;
    h[0] = (_Float16)((v.x - mean) * rinv - bd.x);
    h[1] = (_Float16)((v.y - mean) * rinv - bd.y);
    h[2] = (_Float16)((v.z - mean) * rinv - bd.z);
    h[3] = (_Float16)((v.w - mean) * rinv - bd.w);
    ((v4h*)(A_h + (size_t)row * D_DIM))[tid] = h;
}

// ---------------------------------------------------------------------------
// Kernel 2: W_enc [D,H] fp32 -> Wt [H,D] f16 (transpose + convert, LDS tiled)
// ---------------------------------------------------------------------------
__global__ __launch_bounds__(256)
void wenc_transpose_kernel(const float* __restrict__ W, _Float16* __restrict__ Wt)
{
    __shared__ float tile[32][33];
    const uint32_t tx = threadIdx.x;        // 0..31
    const uint32_t ty = threadIdx.y;        // 0..7
    const uint32_t h0 = blockIdx.x * 32u;
    const uint32_t d0 = blockIdx.y * 32u;
    #pragma unroll
    for (int i = 0; i < 4; ++i)
        tile[ty + 8 * i][tx] = W[(size_t)(d0 + ty + 8 * i) * H_DIM + h0 + tx];
    __syncthreads();
    #pragma unroll
    for (int i = 0; i < 4; ++i)
        Wt[(size_t)(h0 + ty + 8 * i) * D_DIM + d0 + tx] = (_Float16)tile[tx][ty + 8 * i];
}

// ---------------------------------------------------------------------------
// Kernel 3: encode GEMM  pre[B,H] = A_h[B,D] @ Wt[H,D]^T + b_enc
// 128x128 block tile, BK=32, TDM double-buffered LDS, 8 waves x (4x2) WMMA.
// ---------------------------------------------------------------------------
__global__ __launch_bounds__(256, 2)
void encode_gemm_kernel(const _Float16* __restrict__ A,
                        const _Float16* __restrict__ Bt,
                        const float* __restrict__ b_enc,
                        float* __restrict__ pre)
{
    __shared__ __align__(16) _Float16 As[2][128][32];
    __shared__ __align__(16) _Float16 Bs[2][128][32];

    const uint32_t tid   = threadIdx.x;
    const uint32_t lane  = tid & 31u;
    const uint32_t wid   = tid >> 5;
    const uint32_t wm    = wid & 1u;        // 2 waves along M (64 rows each)
    const uint32_t wn    = wid >> 1;        // 4 waves along N (32 cols each)
    const uint32_t bn    = blockIdx.x;      // H / 128
    const uint32_t bm    = blockIdx.y;      // B / 128
    const uint32_t l16   = lane & 15u;
    const uint32_t khalf = (lane < 16u) ? 0u : 8u;   // K offset for upper half-wave

    // Uniform (scalar) predicate: only wave 0 issues TDM ops (TDM ignores EXEC).
    const bool lead = (__builtin_amdgcn_readfirstlane((int)tid) == 0);

    v8f acc[4][2];
    const v8f vzero = {0.f, 0.f, 0.f, 0.f, 0.f, 0.f, 0.f, 0.f};
    #pragma unroll
    for (int mt = 0; mt < 4; ++mt)
        #pragma unroll
        for (int nt = 0; nt < 2; ++nt) acc[mt][nt] = vzero;

    const uint32_t ldsA = (uint32_t)(uintptr_t)&As[0][0][0];  // low 32b = LDS offset
    const uint32_t ldsB = (uint32_t)(uintptr_t)&Bs[0][0][0];
    const _Float16* Ag = A  + (size_t)bm * 128u * D_DIM;
    const _Float16* Bg = Bt + (size_t)bn * 128u * D_DIM;

    if (lead) {
        tdm_load_tile_2d(ldsA, Ag, 32u, 128u, D_DIM);
        tdm_load_tile_2d(ldsB, Bg, 32u, 128u, D_DIM);
        __builtin_amdgcn_s_wait_tensorcnt(0);
    }
    __syncthreads();

    const int KT = D_DIM / 32;
    for (int kt = 0; kt < KT; ++kt) {
        const uint32_t buf = (uint32_t)kt & 1u;
        if (kt + 1 < KT && lead) {
            const uint32_t nb = buf ^ 1u;
            tdm_load_tile_2d(ldsA + nb * (uint32_t)sizeof(As[0]),
                             Ag + (size_t)(kt + 1) * 32, 32u, 128u, D_DIM);
            tdm_load_tile_2d(ldsB + nb * (uint32_t)sizeof(Bs[0]),
                             Bg + (size_t)(kt + 1) * 32, 32u, 128u, D_DIM);
        }

        // Fragment loads: 16-bit 16x32 layout — lanes>=16 offset K by 8,
        // VGPRs 4..7 offset K by 16; two ds_load_b128 per fragment.
        v16h afrag[4], bfrag[2];
        #pragma unroll
        for (int mt = 0; mt < 4; ++mt) {
            const _Float16* s = &As[buf][wm * 64u + (uint32_t)mt * 16u + l16][khalf];
            v8h lo = *(const v8h*)s;
            v8h hi = *(const v8h*)(s + 16);
            afrag[mt] = __builtin_shufflevector(lo, hi, 0,1,2,3,4,5,6,7,8,9,10,11,12,13,14,15);
        }
        #pragma unroll
        for (int nt = 0; nt < 2; ++nt) {
            const _Float16* s = &Bs[buf][wn * 32u + (uint32_t)nt * 16u + l16][khalf];
            v8h lo = *(const v8h*)s;
            v8h hi = *(const v8h*)(s + 16);
            bfrag[nt] = __builtin_shufflevector(lo, hi, 0,1,2,3,4,5,6,7,8,9,10,11,12,13,14,15);
        }

        #pragma unroll
        for (int mt = 0; mt < 4; ++mt)
            #pragma unroll
            for (int nt = 0; nt < 2; ++nt)
                acc[mt][nt] = __builtin_amdgcn_wmma_f32_16x16x32_f16(
                    false, afrag[mt], false, bfrag[nt],
                    (short)0, acc[mt][nt], false, false);

        if (kt + 1 < KT) {
            if (lead) __builtin_amdgcn_s_wait_tensorcnt(0);
            __syncthreads();
        }
    }

    // Epilogue: C layout — VGPR r: lanes 0-15 -> M=r, lanes 16-31 -> M=r+8; N=lane%16
    const uint32_t moff = (lane < 16u) ? 0u : 8u;
    #pragma unroll
    for (int nt = 0; nt < 2; ++nt) {
        const uint32_t gcol = bn * 128u + wn * 32u + (uint32_t)nt * 16u + l16;
        const float bias = b_enc[gcol];
        #pragma unroll
        for (int mt = 0; mt < 4; ++mt) {
            const uint32_t grow = bm * 128u + wm * 64u + (uint32_t)mt * 16u + moff;
            float* dst = pre + (size_t)grow * H_DIM + gcol;
            #pragma unroll
            for (int r = 0; r < 8; ++r)
                dst[(size_t)r * H_DIM] = acc[mt][nt][r] + bias;
        }
    }
}

// ---------------------------------------------------------------------------
// Kernel 4: exact per-row top-64 via 4-pass 8-bit radix select + tie handling
// ---------------------------------------------------------------------------
__device__ __forceinline__ uint32_t f2u(float f)
{
    uint32_t u = __float_as_uint(f);
    return (u & 0x80000000u) ? ~u : (u | 0x80000000u);
}

__global__ __launch_bounds__(256)
void topk_kernel(const float* __restrict__ pre, int* __restrict__ idxOut,
                 float* __restrict__ actOut)
{
    const uint32_t row = blockIdx.x;
    const uint32_t tid = threadIdx.x;
    const float* p = pre + (size_t)row * H_DIM;

    __shared__ unsigned hist[256];
    __shared__ unsigned sh_prefix;
    __shared__ int sh_kth;

    uint32_t prefix = 0, known = 0;
    int kth = TOPK;

    #pragma unroll
    for (int pass = 0; pass < 4; ++pass) {
        const int shift = 24 - 8 * pass;
        hist[tid] = 0;
        __syncthreads();
        for (uint32_t i = tid; i < H_DIM; i += 256u) {
            const uint32_t u = f2u(p[i]);
            if ((u & known) == prefix)
                atomicAdd(&hist[(u >> shift) & 255u], 1u);
        }
        __syncthreads();
        if (tid == 0) {
            int k = kth;
            uint32_t d = 255u;
            while (d > 0u) {
                const int c = (int)hist[d];
                if (c >= k) break;
                k -= c; --d;
            }
            sh_prefix = prefix | (d << shift);
            sh_kth = k;
        }
        __syncthreads();
        prefix = sh_prefix;
        kth = sh_kth;
        known |= (0xFFu << shift);
    }

    __shared__ unsigned eqcnt, wpos;
    if (tid == 0) { eqcnt = 0; wpos = 0; }
    __syncthreads();
    const uint32_t T = prefix;        // exact bits of the k-th largest value
    for (uint32_t i = tid; i < H_DIM; i += 256u) {
        const float f = p[i];
        const uint32_t u = f2u(f);
        bool take = (u > T);
        if (!take && u == T)
            take = (atomicAdd(&eqcnt, 1u) < (unsigned)kth);
        if (take) {
            const unsigned w = atomicAdd(&wpos, 1u);
            idxOut[(size_t)row * TOPK + w] = (int)i;
            actOut[(size_t)row * TOPK + w] = fmaxf(f, 0.0f);   // relu
        }
    }
}

// ---------------------------------------------------------------------------
// Kernel 5: sparse decode + un-normalize + squared-error accumulation
// ---------------------------------------------------------------------------
__global__ __launch_bounds__(256)
void decode_loss_kernel(const float* __restrict__ x, const float* __restrict__ Wdec,
                        const float* __restrict__ b_dec, const int* __restrict__ idx,
                        const float* __restrict__ act, const float* __restrict__ meanArr,
                        const float* __restrict__ stdArr, float* __restrict__ accum)
{
    const uint32_t row = blockIdx.x;
    const uint32_t tid = threadIdx.x;

    __shared__ int   sidx[TOPK];
    __shared__ float sact[TOPK];
    if (tid < TOPK) {
        sidx[tid] = idx[(size_t)row * TOPK + tid];
        sact[tid] = act[(size_t)row * TOPK + tid];
    }
    __syncthreads();

    float4 r = ((const float4*)b_dec)[tid];
    #pragma unroll 8
    for (int j = 0; j < TOPK; ++j) {
        const float a = sact[j];
        const float4 w = ((const float4*)(Wdec + (size_t)sidx[j] * D_DIM))[tid];
        r.x += a * w.x; r.y += a * w.y; r.z += a * w.z; r.w += a * w.w;
    }

    const float mean = meanArr[row], sd = stdArr[row];
    const float4 xv = ((const float4*)(x + (size_t)row * D_DIM))[tid];
    const float d0 = xv.x - (r.x * sd + mean);
    const float d1 = xv.y - (r.y * sd + mean);
    const float d2 = xv.z - (r.z * sd + mean);
    const float d3 = xv.w - (r.w * sd + mean);
    float s = d0 * d0 + d1 * d1 + d2 * d2 + d3 * d3;

    __shared__ float red[256];
    red[tid] = s;
    __syncthreads();
    #pragma unroll
    for (int off = 128; off > 0; off >>= 1) {
        if (tid < (uint32_t)off) red[tid] += red[tid + off];
        __syncthreads();
    }
    if (tid == 0) atomicAdd(accum, red[0]);
}

__global__ void zero_kernel(float* p) { p[0] = 0.0f; }

__global__ void finalize_kernel(const float* __restrict__ accum, float* __restrict__ out)
{
    out[0] = accum[0] * (1.0f / ((float)B_ROWS * (float)D_DIM));
    out[1] = 0.0f;   // aux loss (eval mode)
}

// ---------------------------------------------------------------------------
extern "C" void kernel_launch(void* const* d_in, const int* in_sizes, int n_in,
                              void* d_out, int out_size, void* d_ws, size_t ws_size,
                              hipStream_t stream)
{
    (void)in_sizes; (void)n_in; (void)out_size; (void)ws_size;

    const float* x     = (const float*)d_in[0];   // [B, D]
    const float* W_enc = (const float*)d_in[1];   // [D, H]
    const float* W_dec = (const float*)d_in[2];   // [H, D]
    const float* b_enc = (const float*)d_in[3];   // [H]
    const float* b_dec = (const float*)d_in[4];   // [D]
    // d_in[5] = k (fixed at 64 by the harness)
    float* out = (float*)d_out;

    // Workspace carve (~1.11 GB total; pre dominates at 1 GB)
    char* w = (char*)d_ws;
    auto take = [&](size_t bytes) -> char* {
        char* p = w;
        w += (bytes + 255u) & ~(size_t)255u;
        return p;
    };
    _Float16* A_h     = (_Float16*)take((size_t)B_ROWS * D_DIM * sizeof(_Float16));
    _Float16* Wt      = (_Float16*)take((size_t)H_DIM * D_DIM * sizeof(_Float16));
    float*    pre     = (float*)   take((size_t)B_ROWS * H_DIM * sizeof(float));
    float*    meanArr = (float*)   take((size_t)B_ROWS * sizeof(float));
    float*    stdArr  = (float*)   take((size_t)B_ROWS * sizeof(float));
    int*      idxBuf  = (int*)     take((size_t)B_ROWS * TOPK * sizeof(int));
    float*    actBuf  = (float*)   take((size_t)B_ROWS * TOPK * sizeof(float));
    float*    accum   = (float*)   take(256);

    zero_kernel<<<1, 1, 0, stream>>>(accum);
    layernorm_kernel<<<B_ROWS, 256, 0, stream>>>(x, b_dec, A_h, meanArr, stdArr);
    wenc_transpose_kernel<<<dim3(H_DIM / 32, D_DIM / 32), dim3(32, 8), 0, stream>>>(W_enc, Wt);
    encode_gemm_kernel<<<dim3(H_DIM / 128, B_ROWS / 128), 256, 0, stream>>>(A_h, Wt, b_enc, pre);
    topk_kernel<<<B_ROWS, 256, 0, stream>>>(pre, idxBuf, actBuf);
    decode_loss_kernel<<<B_ROWS, 256, 0, stream>>>(x, W_dec, b_dec, idxBuf, actBuf,
                                                   meanArr, stdArr, accum);
    finalize_kernel<<<1, 1, 0, stream>>>(accum, out);
}
